// SelfAttention_84078279786756
// MI455X (gfx1250) — matrix operand order repaired and checked by reading, not compiled, not run on hardware
//
#include <hip/hip_runtime.h>
#include <hip/hip_bf16.h>

// Problem constants (reference: N=2, L=2048, E=1024, H=16, D=64)
#define NB 2
#define LQ 2048
#define EE 1024
#define HH 16
#define DD 64

typedef __attribute__((ext_vector_type(16))) __bf16 bf16x16;
typedef __attribute__((ext_vector_type(8)))  __bf16 bf16x8;
typedef __attribute__((ext_vector_type(4)))  __bf16 bf16x4;
typedef __attribute__((ext_vector_type(8)))  float  v8f;
typedef __attribute__((ext_vector_type(4)))  float  f32x4;
typedef __attribute__((ext_vector_type(4)))  unsigned int u32x4;
typedef __attribute__((ext_vector_type(8)))  int    i32x8;
typedef __attribute__((ext_vector_type(4)))  int    i32x4;

union ABReg {
    bf16x16 v;
    u32x4   q[2];
};
static_assert(sizeof(bf16x16) == 32, "bf16x16 must be 32 bytes");

__device__ __forceinline__ v8f wmma_bf16(bf16x16 a, bf16x16 b, v8f c) {
    // D = A(16x32 bf16) * B(32x16 bf16) + C(16x16 f32)
    return __builtin_amdgcn_wmma_f32_16x16x32_bf16(
        false, a, false, b, (short)0, c, false, false);
}

// LDS tile geometry: 64 rows padded to 72 bf16 (144 B) to spread banks.
// TDM reproduces this stride via pad_enable (4 DWORD pad per 32 DWORDs).
#define ROWB   144                    // padded row stride in bytes
#define TILEB  (64 * ROWB)            // 9216 B per staged tile
#define OFF_K  0                      // sK[2]
#define OFF_V  (2 * TILEB)            // sVt[2]
#define OFF_P  (4 * TILEB)            // sP[8] (16-row tiles, same 144B stride)
#define PTILEB (16 * ROWB)            // 2304 B per wave P tile
#define SMEM_BYTES (4 * TILEB + 8 * PTILEB)   // 55296 B

#if defined(__HIP_DEVICE_COMPILE__) && __has_builtin(__builtin_amdgcn_tensor_load_to_lds)
#define USE_TDM 1
#else
#define USE_TDM 0
#endif

// ---------------------------------------------------------------------------
// Mask bit-pack: one 32-bit word per wave32 via ballot.  [N,L,L] int32 -> bits
// ---------------------------------------------------------------------------
__global__ __launch_bounds__(256)
void pack_mask_kernel(const int* __restrict__ M, unsigned int* __restrict__ Mw)
{
    const int tid  = threadIdx.x;
    const int lane = tid & 31;
    const size_t word = (size_t)blockIdx.x * 8 + (tid >> 5);
    const int mv = M[word * 32 + lane];
    const unsigned int bits = (unsigned int)__ballot(mv != 0);
    if (lane == 0) Mw[word] = bits;
}

// ---------------------------------------------------------------------------
// fp32 -> bf16 conversion for W_out
// ---------------------------------------------------------------------------
__global__ void cvt_bf16_kernel(const float* __restrict__ src,
                                __bf16* __restrict__ dst, int nElem)
{
    const int i = blockIdx.x * blockDim.x + threadIdx.x;
    if (i < nElem) dst[i] = (__bf16)src[i];
}

// ---------------------------------------------------------------------------
// K/V prep: per-head-packed bf16.
//   Kbf[n,h,key,d]  (row-major, 64-key chunk = 8 KB contiguous rows of 128 B)
//   Vt [n,h,d,key]  (transposed, so B-operand reads are key-contiguous)
// ---------------------------------------------------------------------------
__global__ __launch_bounds__(256)
void cvt_kv_kernel(const float* __restrict__ K, const float* __restrict__ V,
                   __bf16* __restrict__ Kbf, __bf16* __restrict__ Vt)
{
    __shared__ __bf16 sT[64][72];     // padded to dodge bank conflicts

    const int tid = threadIdx.x;
    const int b   = blockIdx.x;       // n*512 + h*32 + kb
    const int kb  = b & 31;
    const int h   = (b >> 5) & 15;
    const int n   = b >> 9;

    const float* Ks = K + (size_t)n * LQ * EE + (size_t)(kb * 64) * EE + h * DD;
    const float* Vs = V + (size_t)n * LQ * EE + (size_t)(kb * 64) * EE + h * DD;
    __bf16* Ko = Kbf + ((size_t)(n * HH + h) * LQ + kb * 64) * DD;
    __bf16* Vo = Vt  +  (size_t)(n * HH + h) * DD * LQ + kb * 64;

    for (int idx = tid; idx < 64 * 16; idx += 256) {
        const int key = idx >> 4;
        const int d   = (idx & 15) * 4;
        f32x4 kf = *(const f32x4*)(Ks + (size_t)key * EE + d);
        bf16x4 k4;
#pragma unroll
        for (int i = 0; i < 4; ++i) k4[i] = (__bf16)kf[i];
        *(bf16x4*)(Ko + key * 64 + d) = k4;
        f32x4 vf = *(const f32x4*)(Vs + (size_t)key * EE + d);
        bf16x4 v4;
#pragma unroll
        for (int i = 0; i < 4; ++i) v4[i] = (__bf16)vf[i];
        *(bf16x4*)&sT[key][d] = v4;
    }
    __syncthreads();
    for (int idx = tid; idx < 64 * 8; idx += 256) {
        const int d  = idx >> 3;
        const int k8 = (idx & 7) * 8;
        bf16x8 o8;
#pragma unroll
        for (int i = 0; i < 8; ++i) o8[i] = sT[k8 + i][d];
        *(bf16x8*)(Vo + (size_t)d * LQ + k8) = o8;
    }
}

#if USE_TDM
// ---------------------------------------------------------------------------
// Tensor Data Mover: load one 64x64 bf16 tile (2D) into LDS with 16B pad per
// 128B line (-> 144B LDS row stride).  D# packing per ISA 8.3/8.4.
// ---------------------------------------------------------------------------
__device__ __forceinline__ void tdm_load_tile(unsigned int lds_addr, const void* g,
                                              unsigned int dim0, unsigned int dim1,
                                              unsigned int stride0)
{
    const unsigned long long ga = (unsigned long long)g;
    u32x4 g0;
    g0[0] = 1u;                                          // count=1 (valid), user mode
    g0[1] = lds_addr;                                    // lds_addr [63:32]
    g0[2] = (unsigned int)ga;                            // global_addr [95:64]
    g0[3] = (unsigned int)((ga >> 32) & 0x1ffffffull)    // global_addr [120:96]
          | (2u << 30);                                  // type=2 (image)
    i32x8 g1;
    g1[0] = (1 << 16)        // data_size = 2 bytes
          | (1 << 20)        // pad_enable
          | (4 << 22)        // pad_interval: 32 DWORDs (128 B)
          | (3 << 25);       // pad_amount:   4 DWORDs (16 B)
    g1[1] = (int)(dim0 << 16);                           // tensor_dim0[15:0] @ 63:48
    g1[2] = (int)((dim0 >> 16) | (dim1 << 16));          // dim0 hi | dim1 lo
    g1[3] = (int)((dim1 >> 16) | (64u << 16));           // dim1 hi | tile_dim0 = 64
    g1[4] = 64;                                          // tile_dim1=64, tile_dim2=0
    g1[5] = (int)stride0;                                // tensor_dim0_stride lo
    g1[6] = 0;                                           // stride0 hi | stride1 lo
    g1[7] = 0;
    const i32x4 z4 = {0, 0, 0, 0};
#if defined(__clang_major__) && (__clang_major__ >= 23)
    const i32x8 z8 = {0, 0, 0, 0, 0, 0, 0, 0};
    __builtin_amdgcn_tensor_load_to_lds(g0, g1, z4, z4, z8, 0);
#else
    __builtin_amdgcn_tensor_load_to_lds(g0, g1, z4, z4, 0);
#endif
}
#else
// ---------------------------------------------------------------------------
// Fallback: per-lane async staging, 512 x 16B slots per tile, 2 per thread.
// ---------------------------------------------------------------------------
__device__ __forceinline__ void issue_async_chunk(const __bf16* gK, const __bf16* gVt,
                                                  unsigned int ldsK, unsigned int ldsV,
                                                  int tid)
{
#pragma unroll
    for (int r = 0; r < 2; ++r) {
        const int slot = tid + r * 256;
        const int row  = slot >> 3;
        const int c16  = slot & 7;
        const unsigned int lk  = ldsK + (unsigned int)(row * ROWB + c16 * 16);
        const unsigned int gko = (unsigned int)(row * 128 + c16 * 16);
        asm volatile("global_load_async_to_lds_b128 %0, %1, %2"
                     :: "v"(lk), "v"(gko), "s"(gK) : "memory");
        const unsigned int lv  = ldsV + (unsigned int)(row * ROWB + c16 * 16);
        const unsigned int gvo = (unsigned int)(row * (LQ * 2) + c16 * 16);
        asm volatile("global_load_async_to_lds_b128 %0, %1, %2"
                     :: "v"(lv), "v"(gvo), "s"(gVt) : "memory");
    }
}
#endif

// Stage one (K, Vt) chunk; TDM path: wave0 -> K tile, wave1 -> Vt tile.
__device__ __forceinline__ void stage_chunk(const __bf16* gK, const __bf16* gVt,
                                            unsigned int ldsK, unsigned int ldsV,
                                            int tid, int wave)
{
#if USE_TDM
    if (wave == 0)      tdm_load_tile(ldsK, gK, 64u, (unsigned)LQ, 64u);
    else if (wave == 1) tdm_load_tile(ldsV, gVt, (unsigned)LQ, 64u, (unsigned)LQ);
    (void)tid;
#else
    issue_async_chunk(gK, gVt, ldsK, ldsV, tid);
    (void)wave;
#endif
}

// Wait until the previous chunk's staging is complete (pending = copies of the
// *next* chunk that may remain in flight).
__device__ __forceinline__ void stage_wait(int wave, bool pending)
{
#if USE_TDM
    if (wave < 2) {
        if (pending) __builtin_amdgcn_s_wait_tensorcnt(1);
        else         __builtin_amdgcn_s_wait_tensorcnt(0);
    }
#else
    (void)wave;
    if (pending) asm volatile("s_wait_asynccnt 4" ::: "memory");
    else         asm volatile("s_wait_asynccnt 0" ::: "memory");
#endif
}

// ---------------------------------------------------------------------------
// Flash attention: one block = (n, h, 128-row q tile), 8 wave32s, each wave
// owns 16 q rows.  Double-buffered TDM/async K/V staging, online softmax,
// output bf16 into workspace [N*L, E] (head h -> cols h*64..h*64+63).
// ---------------------------------------------------------------------------
__global__ __launch_bounds__(256)
void attn_fa_kernel(const float* __restrict__ Q, const __bf16* __restrict__ Kbf,
                    const __bf16* __restrict__ Vt, const unsigned int* __restrict__ Mw,
                    __bf16* __restrict__ O)
{
    __shared__ __align__(128) char smem[SMEM_BYTES];
    const unsigned int lds_base = (unsigned int)(uintptr_t)&smem[0];

    const int tid  = threadIdx.x;
    const int lane = tid & 31;
    const int wave = tid >> 5;
    const int half = lane >> 4;
    const int cl   = lane & 15;

    const int b     = blockIdx.x;
    const int qt    = b & 15;
    const int h     = (b >> 4) & 15;
    const int n     = b >> 8;
    const int qbase = qt * 128;

    const float*  Qh  = Q   + (size_t)n * LQ * EE + (size_t)h * DD;
    const __bf16* Kh  = Kbf + (size_t)(n * HH + h) * LQ * DD;   // [key][64]
    const __bf16* Vth = Vt  + (size_t)(n * HH + h) * DD * LQ;   // [d][2048]

    // ---- Q rows (16 x 64) once, packed into WMMA A layout registers ----
    ABReg aq[2];
    {
        const float* qp = Qh + (size_t)(qbase + wave * 16 + cl) * EE;
#pragma unroll
        for (int t = 0; t < 2; ++t) {
            const int d0 = 32 * t + 8 * half;
            const int d1 = 32 * t + 16 + 8 * half;
            f32x4 f0 = *(const f32x4*)(qp + d0);
            f32x4 f1 = *(const f32x4*)(qp + d0 + 4);
            f32x4 f2 = *(const f32x4*)(qp + d1);
            f32x4 f3 = *(const f32x4*)(qp + d1 + 4);
#pragma unroll
            for (int i = 0; i < 4; ++i) {
                aq[t].v[i]      = (__bf16)f0[i];
                aq[t].v[4 + i]  = (__bf16)f1[i];
                aq[t].v[8 + i]  = (__bf16)f2[i];
                aq[t].v[12 + i] = (__bf16)f3[i];
            }
        }
    }

    v8f o[4];
    float m[8], l[8];
#pragma unroll
    for (int nt = 0; nt < 4; ++nt)
#pragma unroll
        for (int j = 0; j < 8; ++j) o[nt][j] = 0.0f;
#pragma unroll
    for (int j = 0; j < 8; ++j) { m[j] = -3.0e38f; l[j] = 0.0f; }

    const float inv_scale = 1.0f / 32.0f;          // 1/sqrt(E)
    const float NEGM      = -1.0e20f * inv_scale;  // masked score after scale

    __bf16* pw = (__bf16*)(smem + OFF_P + wave * PTILEB);
    const int qrow0 = qbase + wave * 16 + half * 8;
    const unsigned int* mrow = Mw + (size_t)(n * LQ + qrow0) * (LQ / 32);

    // prime buffer 0
    stage_chunk(Kh, Vth, lds_base + OFF_K, lds_base + OFF_V, tid, wave);

    for (int i = 0; i < LQ / 64; ++i) {
        const int kc  = i * 64;
        const int buf = i & 1;

        const bool more = (i + 1 < LQ / 64);
        if (more) {
            const int nb = (i + 1) & 1;
            stage_chunk(Kh + (size_t)(kc + 64) * 64, Vth + (kc + 64),
                        lds_base + OFF_K + nb * TILEB,
                        lds_base + OFF_V + nb * TILEB, tid, wave);
        }
        stage_wait(wave, more);
        __syncthreads();   // chunk i fully resident in LDS for all waves

        const char* sKb = smem + OFF_K + buf * TILEB;
        const char* sVb = smem + OFF_V + buf * TILEB;

        // ---- S = Q * K^T : 4 column tiles x 2 depth steps = 8 WMMAs ----
        v8f c[4];
#pragma unroll
        for (int ct = 0; ct < 4; ++ct) {
#pragma unroll
            for (int j = 0; j < 8; ++j) c[ct][j] = 0.0f;
#pragma unroll
            for (int t = 0; t < 2; ++t) {
                ABReg bk;
                const __bf16* bp = (const __bf16*)(sKb + (ct * 16 + cl) * ROWB) + 32 * t + 8 * half;
                bk.q[0] = *(const u32x4*)bp;
                bk.q[1] = *(const u32x4*)(bp + 16);
                c[ct] = wmma_bf16(aq[t].v, bk.v, c[ct]);
            }
        }

        // ---- mask (bit-packed) + scale ----
        const unsigned int* mc = mrow + (kc >> 5);
#pragma unroll
        for (int j = 0; j < 8; ++j) {
            const unsigned int w0 = mc[j * (LQ / 32)];
            const unsigned int w1 = mc[j * (LQ / 32) + 1];
            c[0][j] = ((w0 >> cl) & 1u)        ? c[0][j] * inv_scale : NEGM;
            c[1][j] = ((w0 >> (16 + cl)) & 1u) ? c[1][j] * inv_scale : NEGM;
            c[2][j] = ((w1 >> cl) & 1u)        ? c[2][j] * inv_scale : NEGM;
            c[3][j] = ((w1 >> (16 + cl)) & 1u) ? c[3][j] * inv_scale : NEGM;
        }

        // ---- online softmax (rows wave-private; reduce across 16-lane half) ----
        float alpha[8];
#pragma unroll
        for (int j = 0; j < 8; ++j) {
            float v0 = fmaxf(fmaxf(c[0][j], c[1][j]), fmaxf(c[2][j], c[3][j]));
#pragma unroll
            for (int off = 1; off < 16; off <<= 1)
                v0 = fmaxf(v0, __shfl_xor(v0, off, 32));
            const float mn = fmaxf(m[j], v0);
            alpha[j] = __expf(m[j] - mn);
            m[j] = mn;
        }
#pragma unroll
        for (int ct = 0; ct < 4; ++ct)
#pragma unroll
            for (int j = 0; j < 8; ++j)
                c[ct][j] = __expf(c[ct][j] - m[j]);
#pragma unroll
        for (int j = 0; j < 8; ++j) {
            float s = (c[0][j] + c[1][j]) + (c[2][j] + c[3][j]);
#pragma unroll
            for (int off = 1; off < 16; off <<= 1)
                s += __shfl_xor(s, off, 32);
            l[j] = l[j] * alpha[j] + s;
        }

        // ---- C-layout -> A-layout via wave-private LDS tile ----
#pragma unroll
        for (int ct = 0; ct < 4; ++ct)
#pragma unroll
            for (int j = 0; j < 8; ++j)
                *((__bf16*)((char*)pw + (half * 8 + j) * ROWB) + ct * 16 + cl) = (__bf16)c[ct][j];

#pragma unroll
        for (int nt = 0; nt < 4; ++nt)
#pragma unroll
            for (int j = 0; j < 8; ++j) o[nt][j] *= alpha[j];

        // ---- O += P * V : 2 depth steps x 4 d tiles = 8 WMMAs ----
#pragma unroll
        for (int t = 0; t < 2; ++t) {
            ABReg ap;
            const __bf16* app = (const __bf16*)((char*)pw + cl * ROWB) + 32 * t + 8 * half;
            ap.q[0] = *(const u32x4*)app;
            ap.q[1] = *(const u32x4*)(app + 16);
#pragma unroll
            for (int nt = 0; nt < 4; ++nt) {
                ABReg bv;
                const __bf16* bvp = (const __bf16*)(sVb + (nt * 16 + cl) * ROWB) + 32 * t + 8 * half;
                bv.q[0] = *(const u32x4*)bvp;
                bv.q[1] = *(const u32x4*)(bvp + 16);
                o[nt] = wmma_bf16(ap.v, bv.v, o[nt]);
            }
        }
        __syncthreads();   // all waves done with buf before it is re-filled
    }

    // ---- normalize + store bf16 attention output ----
    __bf16* oh = O + (size_t)n * LQ * EE + (size_t)h * DD;
#pragma unroll
    for (int j = 0; j < 8; ++j) {
        const float inv = 1.0f / l[j];
        const size_t q = (size_t)(qbase + wave * 16 + half * 8 + j);
#pragma unroll
        for (int nt = 0; nt < 4; ++nt)
            oh[q * EE + nt * 16 + cl] = (__bf16)(o[nt][j] * inv);
    }
}

// ---------------------------------------------------------------------------
// Projection: Out[r, e'] = sum_e attn[r, e] * W[e', e] + b[e']
// Block tile 128x64; wave w -> rows [16w,16w+16), all 64 cols (4 WMMA tiles).
// ---------------------------------------------------------------------------
__global__ __launch_bounds__(256)
void proj_kernel(const __bf16* __restrict__ A, const __bf16* __restrict__ W,
                 const float* __restrict__ bias, float* __restrict__ Out)
{
    __shared__ __bf16 sB[64 * 32];

    const int tid  = threadIdx.x;
    const int lane = tid & 31;
    const int wave = tid >> 5;
    const int half = lane >> 4;
    const int cl   = lane & 15;

    const int rowbase = blockIdx.x * 128 + wave * 16;
    const int colbase = blockIdx.y * 64;

    v8f acc[4];
#pragma unroll
    for (int nt = 0; nt < 4; ++nt)
#pragma unroll
        for (int j = 0; j < 8; ++j) acc[nt][j] = 0.0f;

    for (int kb = 0; kb < EE; kb += 32) {
        __syncthreads();
        {
            const int nloc = tid >> 2;
            const int kloc = (tid & 3) * 8;
            *(u32x4*)&sB[nloc * 32 + kloc] =
                *(const u32x4*)&W[(size_t)(colbase + nloc) * EE + kb + kloc];
        }
        __syncthreads();

        ABReg a;
        const __bf16* ap = A + (size_t)(rowbase + cl) * EE + kb + 8 * half;
        a.q[0] = *(const u32x4*)ap;
        a.q[1] = *(const u32x4*)(ap + 16);

#pragma unroll
        for (int nt = 0; nt < 4; ++nt) {
            ABReg bm;
            const __bf16* bp = &sB[(nt * 16 + cl) * 32 + 8 * half];
            bm.q[0] = *(const u32x4*)bp;
            bm.q[1] = *(const u32x4*)(bp + 16);
            acc[nt] = wmma_bf16(a.v, bm.v, acc[nt]);
        }
    }

#pragma unroll
    for (int nt = 0; nt < 4; ++nt) {
        const int col = colbase + nt * 16 + cl;
        const float bv = bias[col];
#pragma unroll
        for (int j = 0; j < 8; ++j)
            Out[(size_t)(rowbase + half * 8 + j) * EE + col] = acc[nt][j] + bv;
    }
}

// ---------------------------------------------------------------------------
extern "C" void kernel_launch(void* const* d_in, const int* in_sizes, int n_in,
                              void* d_out, int out_size, void* d_ws, size_t ws_size,
                              hipStream_t stream)
{
    (void)in_sizes; (void)n_in; (void)out_size; (void)ws_size;

    const float* Q  = (const float*)d_in[0];
    const float* K  = (const float*)d_in[1];
    const float* V  = (const float*)d_in[2];
    const int*   M  = (const int*)d_in[3];
    const float* W  = (const float*)d_in[4];
    const float* bo = (const float*)d_in[5];
    float* Out      = (float*)d_out;

    // Workspace layout (27 MB total):
    char* ws = (char*)d_ws;
    __bf16*       attn = (__bf16*)ws;                              // 8 MB
    __bf16*       Wbf  = (__bf16*)(ws + 8388608);                  // 2 MB
    __bf16*       Kbf  = (__bf16*)(ws + 10485760);                 // 8 MB
    __bf16*       Vt   = (__bf16*)(ws + 18874368);                 // 8 MB
    unsigned int* Mw   = (unsigned int*)(ws + 27262976);           // 1 MB

    pack_mask_kernel<<<(NB * LQ * LQ / 32) / 8, 256, 0, stream>>>(M, Mw);
    cvt_bf16_kernel<<<(EE * EE + 255) / 256, 256, 0, stream>>>(W, Wbf, EE * EE);
    cvt_kv_kernel<<<NB * HH * (LQ / 64), 256, 0, stream>>>(K, V, Kbf, Vt);
    attn_fa_kernel<<<NB * HH * (LQ / 128), 256, 0, stream>>>(Q, Kbf, Vt, Mw, attn);
    proj_kernel<<<dim3((NB * LQ) / 128, EE / 64), 256, 0, stream>>>(attn, Wbf, bo, Out);
}
